// MultiHeadAttention_69638599737373
// MI455X (gfx1250) — matrix operand order repaired and checked by reading, not compiled, not run on hardware
//
#include <hip/hip_runtime.h>

#define DEV __device__ __forceinline__

typedef __attribute__((ext_vector_type(16))) _Float16 v16h;
typedef __attribute__((ext_vector_type(8)))  _Float16 v8h;
typedef __attribute__((ext_vector_type(2)))  _Float16 h2;
typedef __attribute__((ext_vector_type(8)))  float    v8f;
typedef __attribute__((ext_vector_type(2)))  float    f2;

constexpr int Bn = 4, Tn = 2048, Cn = 1024, Hn = 16, Dn = 64;
constexpr int Mn = Bn * Tn;                 // 8192 token rows
constexpr size_t CC = (size_t)Cn * Cn;      // elems per weight matrix

union AFrag { v16h v; h2 p[8]; _Float16 h[16]; };
union BFrag { v16h v; v8h q[2]; };
union CFrag { v8f v; float f[8]; };

DEV v8f wmma_f16(v16h a, v16h b, v8f c) {
  // D(16x16,f32) = A(16x32,f16) * B(32x16,f16) + C
  return __builtin_amdgcn_wmma_f32_16x16x32_f16(false, a, false, b, (short)0, c,
                                                false, false);
}

// A-fragment (16x32 f16, row-major source, lda in elements).
// Lane = row (lane&15); K pairs: vgpr i -> k = koff + 2i{,+1}, vgpr i+4 -> +16,
// with koff = 8 for lanes 16..31 (ISA 7.12.2 16-bit A layout).
DEV v16h load_a_f16(const _Float16* A, int lda, int row0, int k0) {
  const int lane = threadIdx.x & 31;
  const _Float16* p = A + (size_t)(row0 + (lane & 15)) * lda + (k0 + ((lane >> 4) << 3));
  AFrag a;
#pragma unroll
  for (int i = 0; i < 4; ++i) {
    a.p[i]     = *(const h2*)(p + 2 * i);
    a.p[i + 4] = *(const h2*)(p + 16 + 2 * i);
  }
  return a.v;
}

// Same A-fragment but converting from an f32 row-major source on the fly.
DEV v16h load_a_f32(const float* A, int lda, int row0, int k0) {
  const int lane = threadIdx.x & 31;
  const float* p = A + (size_t)(row0 + (lane & 15)) * lda + (k0 + ((lane >> 4) << 3));
  AFrag a;
#pragma unroll
  for (int i = 0; i < 4; ++i) {
    f2 x = *(const f2*)(p + 2 * i);
    f2 y = *(const f2*)(p + 16 + 2 * i);
    h2 lo = { (_Float16)x.x, (_Float16)x.y };
    h2 hi = { (_Float16)y.x, (_Float16)y.y };
    a.p[i]     = lo;
    a.p[i + 4] = hi;
  }
  return a.v;
}

// B-fragment (32x16 f16). Source "Bt" is N-major: row n holds the 16..ldb K
// values of output column n. Lane = column (lane&15); lane holds K halves
// k0 + 8*(lane>>4) + {0..7} and +16 + {0..7}  -> two 16-byte loads.
DEV v16h load_b_f16(const _Float16* Bt, int ldb, int n0, int k0) {
  const int lane = threadIdx.x & 31;
  const _Float16* p = Bt + (size_t)(n0 + (lane & 15)) * ldb + (k0 + ((lane >> 4) << 3));
  BFrag b;
  b.q[0] = *(const v8h*)(p);
  b.q[1] = *(const v8h*)(p + 16);
  return b.v;
}

// ---- Kernel 0: weights f32 -> transposed f16 -------------------------------
__global__ void convert_weights_k(const float* __restrict__ Wq, const float* __restrict__ Wk,
                                  const float* __restrict__ Wv, const float* __restrict__ Wo,
                                  _Float16* __restrict__ dst) {
  const int idx = blockIdx.x * blockDim.x + threadIdx.x;   // n*C + k
  const int w = blockIdx.y;
  const float* src = (w == 0) ? Wq : (w == 1) ? Wk : (w == 2) ? Wv : Wo;
  const int n = idx >> 10;
  const int k = idx & (Cn - 1);
  dst[(size_t)w * CC + idx] = (_Float16)src[(size_t)k * Cn + n];   // dst[n][k] = src[k][n]
}

// ---- Kernel 1: fused QKV projection ---------------------------------------
// Wave computes a 16x64 tile of all three projections (12 WMMAs / K-step).
// Q is pre-scaled by 1/sqrt(D); V is stored transposed (B,H,D,T).
__global__ void __launch_bounds__(128) qkv_gemm_k(const float* __restrict__ X,
                                                  const _Float16* __restrict__ WqT,
                                                  const _Float16* __restrict__ WkT,
                                                  const _Float16* __restrict__ WvT,
                                                  _Float16* __restrict__ Qh,
                                                  _Float16* __restrict__ Kh,
                                                  _Float16* __restrict__ Vt) {
  const int lane = threadIdx.x & 31;
  const int wave = threadIdx.x >> 5;
  const int m0 = blockIdx.x * 64 + wave * 16;
  const int n0 = blockIdx.y * 64;

  CFrag acc[3][4];
#pragma unroll
  for (int w = 0; w < 3; ++w)
#pragma unroll
    for (int nt = 0; nt < 4; ++nt)
#pragma unroll
      for (int r = 0; r < 8; ++r) acc[w][nt].f[r] = 0.f;

  for (int kk = 0; kk < Cn; kk += 32) {
    const v16h a = load_a_f32(X, Cn, m0, kk);
#pragma unroll
    for (int nt = 0; nt < 4; ++nt) {
      acc[0][nt].v = wmma_f16(a, load_b_f16(WqT, Cn, n0 + nt * 16, kk), acc[0][nt].v);
      acc[1][nt].v = wmma_f16(a, load_b_f16(WkT, Cn, n0 + nt * 16, kk), acc[1][nt].v);
      acc[2][nt].v = wmma_f16(a, load_b_f16(WvT, Cn, n0 + nt * 16, kk), acc[2][nt].v);
    }
  }

  const int h = blockIdx.y;               // 64-wide n-tile == one head
  const int rowoff = (lane >> 4) << 3;
#pragma unroll
  for (int nt = 0; nt < 4; ++nt) {
    const int d = nt * 16 + (lane & 15);
#pragma unroll
    for (int r = 0; r < 8; ++r) {
      const int m = m0 + rowoff + r;
      const int b = m >> 11;              // / T
      const int t = m & (Tn - 1);
      const size_t qk = (((size_t)b * Hn + h) * Tn + t) * Dn + d;
      Qh[qk] = (_Float16)(acc[0][nt].f[r] * 0.125f);   // 1/sqrt(64)
      Kh[qk] = (_Float16)acc[1][nt].f[r];
      Vt[(((size_t)b * Hn + h) * Dn + d) * Tn + t] = (_Float16)acc[2][nt].f[r];
    }
  }
}

// ---- Kernel 2: flash attention (one wave = 16 query rows) ------------------
__global__ void __launch_bounds__(128) attn_k(const _Float16* __restrict__ Qh,
                                              const _Float16* __restrict__ Kh,
                                              const _Float16* __restrict__ Vt,
                                              _Float16* __restrict__ ctxh) {
  __shared__ _Float16 sP[4][16 * 32];     // per-wave P tile (C-layout -> A-layout staging)
  const int lane = threadIdx.x & 31;
  const int wave = threadIdx.x >> 5;
  const int q0 = blockIdx.x * 64 + wave * 16;
  const int bh = blockIdx.y;

  const _Float16* Qp = Qh + (size_t)bh * Tn * Dn;
  const _Float16* Kp = Kh + (size_t)bh * Tn * Dn;
  const _Float16* Vp = Vt + (size_t)bh * Dn * Tn;

  const v16h aq0 = load_a_f16(Qp, Dn, q0, 0);
  const v16h aq1 = load_a_f16(Qp, Dn, q0, 32);

  CFrag o[4];
  float mrow[8], lrow[8];
#pragma unroll
  for (int r = 0; r < 8; ++r) { mrow[r] = -__builtin_inff(); lrow[r] = 0.f; }
#pragma unroll
  for (int c = 0; c < 4; ++c)
#pragma unroll
    for (int r = 0; r < 8; ++r) o[c].f[r] = 0.f;

  const int rowbase = q0 + ((lane >> 4) << 3);
  const int nkb = (q0 + 47) >> 5;         // 32-key blocks up to the diagonal
  for (int kb = 0; kb < nkb; ++kb) {
    const int k0 = kb * 32;

    // S(16x32) = Q(16x64) * K^T(64x32), two 16-wide column tiles
    CFrag s[2];
#pragma unroll
    for (int t = 0; t < 2; ++t) {
#pragma unroll
      for (int r = 0; r < 8; ++r) s[t].f[r] = 0.f;
      s[t].v = wmma_f16(aq0, load_b_f16(Kp, Dn, k0 + t * 16, 0), s[t].v);
      s[t].v = wmma_f16(aq1, load_b_f16(Kp, Dn, k0 + t * 16, 32), s[t].v);
    }

    if (k0 + 31 > q0) {                   // diagonal block: causal mask
#pragma unroll
      for (int t = 0; t < 2; ++t) {
        const int col = k0 + t * 16 + (lane & 15);
#pragma unroll
        for (int r = 0; r < 8; ++r)
          if (col > rowbase + r) s[t].f[r] = -__builtin_inff();
      }
    }

    // online softmax: row stats live replicated across each 16-lane half
#pragma unroll
    for (int r = 0; r < 8; ++r) {
      float mx = fmaxf(s[0].f[r], s[1].f[r]);
#pragma unroll
      for (int sh = 1; sh <= 8; sh <<= 1) mx = fmaxf(mx, __shfl_xor(mx, sh, 32));
      const float mnew = fmaxf(mrow[r], mx);
      const float alpha = __expf(mrow[r] - mnew);
      mrow[r] = mnew;
      const float p0 = __expf(s[0].f[r] - mnew);
      const float p1 = __expf(s[1].f[r] - mnew);
      s[0].f[r] = p0; s[1].f[r] = p1;
      float ps = p0 + p1;
#pragma unroll
      for (int sh = 1; sh <= 8; sh <<= 1) ps += __shfl_xor(ps, sh, 32);
      lrow[r] = lrow[r] * alpha + ps;
#pragma unroll
      for (int c = 0; c < 4; ++c) o[c].f[r] *= alpha;
    }

    // C-layout -> A-layout transpose of P through LDS (per-wave tile)
    _Float16* sp = sP[wave];
#pragma unroll
    for (int t = 0; t < 2; ++t)
#pragma unroll
      for (int r = 0; r < 8; ++r)
        sp[(((lane >> 4) << 3) + r) * 32 + t * 16 + (lane & 15)] = (_Float16)s[t].f[r];
    asm volatile("s_wait_dscnt 0x0" ::: "memory");   // LDS stores visible to our loads
    const v16h ap = load_a_f16(sp, 32, 0, 0);

    // O(16x64) += P(16x32) * V(32x64); V^T layout makes B-frags contiguous
#pragma unroll
    for (int c = 0; c < 4; ++c)
      o[c].v = wmma_f16(ap, load_b_f16(Vp, Tn, c * 16, k0), o[c].v);
  }

  float invl[8];
#pragma unroll
  for (int r = 0; r < 8; ++r) invl[r] = 1.f / lrow[r];
  const int b = bh >> 4, h = bh & 15;
#pragma unroll
  for (int c = 0; c < 4; ++c) {
    const int d = h * Dn + c * 16 + (lane & 15);
#pragma unroll
    for (int r = 0; r < 8; ++r) {
      const int t = q0 + ((lane >> 4) << 3) + r;
      ctxh[((size_t)b * Tn + t) * Cn + d] = (_Float16)(o[c].f[r] * invl[r]);
    }
  }
}

// ---- Kernel 3: output projection + bias (f32 out) --------------------------
__global__ void __launch_bounds__(128) out_gemm_k(const _Float16* __restrict__ ctxh,
                                                  const _Float16* __restrict__ WoT,
                                                  const float* __restrict__ bo,
                                                  float* __restrict__ out) {
  const int lane = threadIdx.x & 31;
  const int wave = threadIdx.x >> 5;
  const int m0 = blockIdx.x * 64 + wave * 16;
  const int n0 = blockIdx.y * 64;

  CFrag acc[4];
#pragma unroll
  for (int nt = 0; nt < 4; ++nt)
#pragma unroll
    for (int r = 0; r < 8; ++r) acc[nt].f[r] = 0.f;

  for (int kk = 0; kk < Cn; kk += 32) {
    const v16h a = load_a_f16(ctxh, Cn, m0, kk);
#pragma unroll
    for (int nt = 0; nt < 4; ++nt)
      acc[nt].v = wmma_f16(a, load_b_f16(WoT, Cn, n0 + nt * 16, kk), acc[nt].v);
  }

#pragma unroll
  for (int nt = 0; nt < 4; ++nt) {
    const int n = n0 + nt * 16 + (lane & 15);
    const float bias = bo[n];
#pragma unroll
    for (int r = 0; r < 8; ++r) {
      const int m = m0 + ((lane >> 4) << 3) + r;
      out[(size_t)m * Cn + n] = acc[nt].f[r] + bias;
    }
  }
}

extern "C" void kernel_launch(void* const* d_in, const int* in_sizes, int n_in,
                              void* d_out, int out_size, void* d_ws, size_t ws_size,
                              hipStream_t stream) {
  const float* x  = (const float*)d_in[0];
  const float* Wq = (const float*)d_in[1];
  const float* Wk = (const float*)d_in[2];
  const float* Wv = (const float*)d_in[3];
  const float* Wo = (const float*)d_in[4];
  const float* bo = (const float*)d_in[5];
  float* out = (float*)d_out;

  const size_t wBytes   = 4 * CC * sizeof(_Float16);           // 8 MB  (WqT,WkT,WvT,WoT)
  const size_t matBytes = (size_t)Mn * Cn * sizeof(_Float16);  // 16 MB (Q / K / V^T / ctx)
  if (ws_size < wBytes + 4 * matBytes) return;                 // 72 MB total

  char* ws = (char*)d_ws;
  _Float16* wT  = (_Float16*)ws;
  _Float16* Qh  = (_Float16*)(ws + wBytes);
  _Float16* Kh  = (_Float16*)(ws + wBytes + matBytes);
  _Float16* Vt  = (_Float16*)(ws + wBytes + 2 * matBytes);
  _Float16* ctx = (_Float16*)(ws + wBytes + 3 * matBytes);

  convert_weights_k<<<dim3((unsigned)(CC / 256), 4), 256, 0, stream>>>(Wq, Wk, Wv, Wo, wT);
  qkv_gemm_k<<<dim3(Mn / 64, Cn / 64), 128, 0, stream>>>(x, wT, wT + CC, wT + 2 * CC, Qh, Kh, Vt);
  attn_k<<<dim3(Tn / 64, Bn * Hn), 128, 0, stream>>>(Qh, Kh, Vt, ctx);
  out_gemm_k<<<dim3(Mn / 64, Cn / 64), 128, 0, stream>>>(ctx, wT + 3 * CC, bo, out);
}
